// Net_46308337385577
// MI455X (gfx1250) — matrix operand order repaired and checked by reading, not compiled, not run on hardware
//
#include <hip/hip_runtime.h>
#include <hip/hip_bf16.h>

#define NN 38400
#define EE 300000
#define GG 512

typedef __attribute__((ext_vector_type(16))) __bf16 v16bf;
typedef __attribute__((ext_vector_type(8)))  float  v8f;

union V16 { v16bf v; uint4 q[2]; };
union B8  { __bf16 b[8]; uint4 q; };

__device__ __forceinline__ float elu_f(float x) {
    return x > 0.f ? x : (expf(x) - 1.f);
}

// ---------------------------------------------------------------- zero scratch
__global__ void k_zero(float* __restrict__ p, int n) {
    int i = blockIdx.x * blockDim.x + threadIdx.x;
    if (i < n) p[i] = 0.f;
}

// ---------------------------------------------------------------- layer-1 edges
// msg[e,o] = sum_k (x[src]*relu(ea@w1+b1))[k] * w2[k,o] + x[src]*b2[o]
// One wave per 16-edge tile; K=32 -> exactly one WMMA per 16x16 C tile.
__global__ __launch_bounds__(256) void k_l1_edges(
    const float* __restrict__ xg, const int* __restrict__ ei,
    const float* __restrict__ eag,
    const float* __restrict__ w1g, const float* __restrict__ b1g,
    const float* __restrict__ w2g, const float* __restrict__ b2g,
    float* __restrict__ agg1)
{
    // B fragments for w2[32,32] in the same k order as A: elem j <-> K = 16*h + j
    __shared__ uint4 lds1[128];            // [t(2)][j8(2)][L(32)] 16B records
    __shared__ float s_w10[32], s_w11[32], s_b1[32], s_b2[32];
    int tid = threadIdx.x;
    if (tid < 128) {
        int t = tid >> 6, j8 = (tid >> 5) & 1, L = tid & 31;
        int h = L >> 4, n = L & 15, o = t * 16 + n;
        B8 pk;
#pragma unroll
        for (int jp = 0; jp < 8; ++jp) {
            int K = 16 * h + 8 * j8 + jp;
            pk.b[jp] = (__bf16)w2g[K * 32 + o];
        }
        lds1[tid] = pk.q;
    }
    if (tid < 32) {
        s_w10[tid] = w1g[tid]; s_w11[tid] = w1g[32 + tid];
        s_b1[tid] = b1g[tid];  s_b2[tid] = b2g[tid];
    }
    __syncthreads();

    int wave = tid >> 5, L = tid & 31, m = L & 15, h = L >> 4, n = m;
    int tile = blockIdx.x * 8 + wave;
    if (tile >= 18750) return;
    int e = tile * 16 + m;
    int src = ei[e], dst = ei[EE + e];
    float ea0 = eag[2 * e], ea1 = eag[2 * e + 1];
    float xs = xg[src];

    // A fragment: elem j <-> K = 16*h + j
    V16 av;
#pragma unroll
    for (int j = 0; j < 16; ++j) {
        int kk = j + 16 * h;
        float v = fmaxf(ea0 * s_w10[kk] + ea1 * s_w11[kk] + s_b1[kk], 0.f) * xs;
        av.v[j] = (__bf16)v;
    }
    V16 bt0, bt1;
    bt0.q[0] = lds1[L];      bt0.q[1] = lds1[32 + L];
    bt1.q[0] = lds1[64 + L]; bt1.q[1] = lds1[96 + L];

    v8f c0 = {}; v8f c1 = {};
    c0 = __builtin_amdgcn_wmma_f32_16x16x32_bf16(false, av.v, false, bt0.v, (short)0, c0, false, false);
    c1 = __builtin_amdgcn_wmma_f32_16x16x32_bf16(false, av.v, false, bt1.v, (short)0, c1, false, false);

#pragma unroll
    for (int r = 0; r < 8; ++r) {
        int mi = r + 8 * h;                       // C row -> edge index in tile
        int   dstm = __shfl(dst, mi, 32);
        float xsm  = __shfl(xs, mi, 32);
        atomicAdd(&agg1[dstm * 32 + n],       c0[r] + xsm * s_b2[n]);
        atomicAdd(&agg1[dstm * 32 + 16 + n],  c1[r] + xsm * s_b2[16 + n]);
    }
}

// ---------------------------------------------------------------- layer-1 nodes
__global__ void k_l1_nodes(const float* __restrict__ xg, const float* __restrict__ agg1,
                           const float* __restrict__ root1, const float* __restrict__ bias1,
                           float* __restrict__ h1f, __bf16* __restrict__ h1bf)
{
    int i = blockIdx.x * blockDim.x + threadIdx.x;
    if (i >= NN * 32) return;
    int v = i >> 5, o = i & 31;
    float h = elu_f(xg[v] * root1[o] + agg1[i] + bias1[o]);
    h1f[i] = h;
    h1bf[i] = (__bf16)h;
}

// ------------------------------------------- scatter Sum_{e:dst=v} h1[src[e],:]
// (carries the factored-out nn2_b2 contribution of layer 2)
__global__ void k_scatter_s(const int* __restrict__ ei, const float* __restrict__ h1f,
                            float* __restrict__ Ssum)
{
    int idx = blockIdx.x * blockDim.x + threadIdx.x;
    if (idx >= EE * 32) return;
    int e = idx >> 5, i = idx & 31;
    int src = ei[e], dst = ei[EE + e];
    atomicAdd(&Ssum[dst * 32 + i], h1f[src * 32 + i]);
}

// ---------------------------------------------------------------- layer-2 edges
// Single GEMM M=E, K=2048 (pairs (k,i)), N=64 with rank-1 on-the-fly A tiles.
// K split 4-ways across blocks; each block keeps its 16-chunk B slice (64KB)
// in LDS in exact B-fragment order (conflict-free b128 reads).
// Dual accumulators break the WMMA D->C serial chain; next-tile edge data is
// prefetched (global_prefetch_b8) behind the 16-WMMA compute phase.
__global__ __launch_bounds__(256) void k_l2_edges(
    const int* __restrict__ ei, const float* __restrict__ eag,
    const __bf16* __restrict__ h1bf,
    const float* __restrict__ w1g,   // nn2_w1 [2,64]
    const float* __restrict__ b1g,   // nn2_b1 [64]
    const float* __restrict__ w2g,   // nn2_w2 [64,2048]
    float* __restrict__ agg2)
{
    __shared__ uint4 ldsB[4096];     // 16 chunks * 4 ntiles * 2 j8 * 32 lanes = 64KB
    int tid    = threadIdx.x;
    int kphase = blockIdx.x & 3;
    int group  = blockIdx.x >> 2;
    int NG     = gridDim.x >> 2;
    int kbase  = kphase * 16;

    // Fill B slice: record (c,t,j8,L), elem jp -> i = 16*h + 8*j8 + jp, o = t*16 + n
    for (int rec = tid; rec < 4096; rec += 256) {
        int L = rec & 31, j8 = (rec >> 5) & 1, t = (rec >> 6) & 3, c = rec >> 8;
        int h = L >> 4, n = L & 15, o = t * 16 + n, k = kbase + c;
        B8 pk;
#pragma unroll
        for (int jp = 0; jp < 8; ++jp) {
            int i = 16 * h + 8 * j8 + jp;
            pk.b[jp] = (__bf16)w2g[k * 2048 + i * 64 + o];
        }
        ldsB[rec] = pk.q;
    }
    __syncthreads();

    int wave = tid >> 5, L = tid & 31;
    int s = wave >> 2, t = wave & 3;          // 2 edge-subtiles x 4 n-tiles
    int m = L & 15, h = L >> 4, n = m;

    for (int tile = group; tile < 9375; tile += NG) {   // 9375 * 32 = E exactly
        int e = tile * 32 + s * 16 + m;
        int src = ei[e], dst = ei[EE + e];
        float ea0 = eag[2 * e], ea1 = eag[2 * e + 1];

        // Prefetch next tile's edge indices / attributes (hidden behind WMMAs).
        int tnext = tile + NG;
        if (tnext < 9375) {
            int en = tnext * 32 + s * 16 + m;
            __builtin_prefetch(&ei[en], 0, 1);
            __builtin_prefetch(&ei[EE + en], 0, 1);
            __builtin_prefetch(&eag[2 * en], 0, 1);
        }

        // Gather h1[src,:] directly in A-fragment i-order:
        // elems 0..15 -> i = 16h..16h+15
        const uint4* hp = (const uint4*)(h1bf + src * 32);
        V16 va;
        va.q[0] = hp[2 * h];
        va.q[1] = hp[2 * h + 1];

        v8f acc0 = {}; v8f acc1 = {};
#pragma unroll
        for (int c = 0; c < 16; ++c) {
            int k = kbase + c;                               // uniform -> s_load
            float hs = fmaxf(ea0 * w1g[k] + ea1 * w1g[64 + k] + b1g[k], 0.f);
            __bf16 hb = (__bf16)hs;
            V16 av;
#pragma unroll
            for (int j = 0; j < 16; ++j) av.v[j] = hb * va.v[j];  // rank-1 A tile
            V16 bb;
            int base = c * 256 + t * 64 + L;
            bb.q[0] = ldsB[base];
            bb.q[1] = ldsB[base + 32];
            if (c & 1)
                acc1 = __builtin_amdgcn_wmma_f32_16x16x32_bf16(false, av.v, false, bb.v,
                                                               (short)0, acc1, false, false);
            else
                acc0 = __builtin_amdgcn_wmma_f32_16x16x32_bf16(false, av.v, false, bb.v,
                                                               (short)0, acc0, false, false);
        }
#pragma unroll
        for (int r = 0; r < 8; ++r) {
            int dstm = __shfl(dst, r + 8 * h, 32);
            atomicAdd(&agg2[dstm * 64 + t * 16 + n], acc0[r] + acc1[r]);
        }
    }
}

// ---------------------------------------------------------------- layer-2 nodes + pool
__global__ void k_l2_nodes(const float* __restrict__ h1f, const float* __restrict__ Ssum,
                           const float* __restrict__ agg2,
                           const float* __restrict__ root2, const float* __restrict__ bias2,
                           const float* __restrict__ b2g,   // nn2_b2 [2048] = B2[i][o]
                           const int* __restrict__ batch,
                           float* __restrict__ psum, float* __restrict__ pcnt)
{
    int idx = blockIdx.x * blockDim.x + threadIdx.x;
    if (idx >= NN * 64) return;
    int v = idx >> 6, o = idx & 63;
    float acc = agg2[idx] + bias2[o];
    const float* hv = h1f + v * 32;
    const float* sv = Ssum + v * 32;
#pragma unroll 8
    for (int k = 0; k < 32; ++k) acc += hv[k] * root2[k * 64 + o];
#pragma unroll 8
    for (int i = 0; i < 32; ++i) acc += sv[i] * b2g[i * 64 + o];
    float h2 = elu_f(acc);
    int b = batch[v];
    atomicAdd(&psum[b * 64 + o], h2);
    if (o == 0) atomicAdd(&pcnt[b], 1.f);
}

// ---------------------------------------------------------------- head
__global__ __launch_bounds__(128) void k_head(
    const float* __restrict__ psum, const float* __restrict__ pcnt,
    const float* __restrict__ l1w, const float* __restrict__ l1b,
    const float* __restrict__ l2w, const float* __restrict__ l2b,
    float* __restrict__ out)
{
    __shared__ float gv[64], y1[128], zz[10], s_lse;
    int g = blockIdx.x, t = threadIdx.x;
    if (t < 64) gv[t] = psum[g * 64 + t] / fmaxf(pcnt[g], 1.f);
    __syncthreads();
    {
        float s = l1b[t];
        for (int k = 0; k < 64; ++k) s += gv[k] * l1w[k * 128 + t];
        y1[t] = elu_f(s);
    }
    __syncthreads();
    if (t < 10) {
        float s = l2b[t];
        for (int j = 0; j < 128; ++j) s += y1[j] * l2w[j * 10 + t];
        zz[t] = s;
    }
    __syncthreads();
    if (t == 0) {
        float mx = zz[0];
        for (int i = 1; i < 10; ++i) mx = fmaxf(mx, zz[i]);
        float ss = 0.f;
        for (int i = 0; i < 10; ++i) ss += expf(zz[i] - mx);
        s_lse = mx + logf(ss);
    }
    __syncthreads();
    if (t < 10) out[g * 10 + t] = zz[t] - s_lse;
}

// ----------------------------------------------------------------
extern "C" void kernel_launch(void* const* d_in, const int* in_sizes, int n_in,
                              void* d_out, int out_size, void* d_ws, size_t ws_size,
                              hipStream_t stream)
{
    (void)in_sizes; (void)n_in; (void)out_size; (void)ws_size;
    const float* x     = (const float*)d_in[0];
    const int*   ei    = (const int*)d_in[1];
    const float* ea    = (const float*)d_in[2];
    const int*   batch = (const int*)d_in[3];
    const float* n1w1  = (const float*)d_in[4];
    const float* n1b1  = (const float*)d_in[5];
    const float* n1w2  = (const float*)d_in[6];
    const float* n1b2  = (const float*)d_in[7];
    const float* root1 = (const float*)d_in[8];
    const float* bias1 = (const float*)d_in[9];
    const float* n2w1  = (const float*)d_in[10];
    const float* n2b1  = (const float*)d_in[11];
    const float* n2w2  = (const float*)d_in[12];
    const float* n2b2  = (const float*)d_in[13];
    const float* root2 = (const float*)d_in[14];
    const float* bias2 = (const float*)d_in[15];
    const float* l1w   = (const float*)d_in[16];
    const float* l1b   = (const float*)d_in[17];
    const float* l2w   = (const float*)d_in[18];
    const float* l2b   = (const float*)d_in[19];
    float* out = (float*)d_out;

    float* ws   = (float*)d_ws;
    float* agg1 = ws;                    // N*32
    float* Ssum = agg1 + NN * 32;        // N*32
    float* agg2 = Ssum + NN * 32;        // N*64
    float* psum = agg2 + NN * 64;        // G*64
    float* pcnt = psum + GG * 64;        // G
    float* h1f  = pcnt + GG;             // N*32
    __bf16* h1bf = (__bf16*)(h1f + NN * 32); // N*32 bf16

    int nz = NN * 128 + GG * 65;         // agg1+Ssum+agg2+psum+pcnt
    k_zero<<<(nz + 255) / 256, 256, 0, stream>>>(ws, nz);
    k_l1_edges<<<2344, 256, 0, stream>>>(x, ei, ea, n1w1, n1b1, n1w2, n1b2, agg1);
    k_l1_nodes<<<(NN * 32 + 255) / 256, 256, 0, stream>>>(x, agg1, root1, bias1, h1f, h1bf);
    k_scatter_s<<<(EE * 32 + 255) / 256, 256, 0, stream>>>(ei, h1f, Ssum);
    k_l2_edges<<<1024, 256, 0, stream>>>(ei, ea, h1bf, n2w1, n2b1, n2w2, agg2);
    k_l2_nodes<<<(NN * 64 + 255) / 256, 256, 0, stream>>>(h1f, Ssum, agg2, root2, bias2,
                                                          n2b2, batch, psum, pcnt);
    k_head<<<GG, 128, 0, stream>>>(psum, pcnt, l1w, l1b, l2w, l2b, out);
}
